// HighwayLSTM_37383395344581
// MI455X (gfx1250) — matrix operand order repaired and checked by reading, not compiled, not run on hardware
//
#include <hip/hip_runtime.h>

typedef __attribute__((ext_vector_type(8)))  float  v8f;
typedef __attribute__((ext_vector_type(8)))  __bf16 v8bf;
typedef __attribute__((ext_vector_type(16))) __bf16 v16bf;

#define NB 64            // batch
#define NT 256           // time steps
#define ND 200           // input dim (layer 0)
#define NH 300           // hidden dim
#define NL 8             // layers
#define HP 304           // hidden padded to 16
#define KPH 320          // K padded to 32 (hidden)
#define KPD 224          // K padded to 32 (input dim)
#define NG6 (6*HP)       // 1824 padded columns of W_ih (6 gates incl highway)
#define NG5 (5*HP)       // 1520 padded columns of W_hh (5 gates)
#define MR  (NT*NB)      // 16384 GEMM rows
#define RBLK (HP/16)     // 19 recurrence workgroups
#define LPAD 8           // LDS row padding (elems) -> conflict-free b128 reads

// ---------- helpers ----------
__device__ __forceinline__ unsigned short f2bf(float f){
  union { float f; unsigned u; } v; v.f = f;
  unsigned u = v.u;
  u += 0x7fffu + ((u >> 16) & 1u);     // round-to-nearest-even
  return (unsigned short)(u >> 16);
}
__device__ __forceinline__ float sigm(float x){ return 1.f/(1.f+__expf(-x)); }

__device__ __forceinline__ v16bf cat8(v8bf lo, v8bf hi){
  return __builtin_shufflevector(lo, hi, 0,1,2,3,4,5,6,7,8,9,10,11,12,13,14,15);
}

// ---------- weight repack (f32 -> bf16, transposed [N][K], padded) ----------
__global__ void pack_wih0_k(const float* __restrict__ W, unsigned short* __restrict__ dst){
  const int total = NG6*KPD;
  for (int i = blockIdx.x*blockDim.x + threadIdx.x; i < total; i += gridDim.x*blockDim.x){
    int k = i % KPD, n = i / KPD;
    int g = n / HP, j = n % HP;
    float v = 0.f;
    if (j < NH && k < ND) v = W[(size_t)k*(6*NH) + g*NH + j];   // [200,1800]
    dst[i] = f2bf(v);
  }
}
__global__ void pack_wihr_k(const float* __restrict__ W, unsigned short* __restrict__ dst){
  const int per = NG6*KPH;
  const int total = 7*per;
  for (int i = blockIdx.x*blockDim.x + threadIdx.x; i < total; i += gridDim.x*blockDim.x){
    int l = i / per, r = i % per;
    int k = r % KPH, n = r / KPH;
    int g = n / HP, j = n % HP;
    float v = 0.f;
    if (j < NH && k < NH) v = W[(size_t)l*NH*(6*NH) + (size_t)k*(6*NH) + g*NH + j]; // [7,300,1800]
    dst[i] = f2bf(v);
  }
}
__global__ void pack_whh_k(const float* __restrict__ W, unsigned short* __restrict__ dst){
  const int per = NG5*KPH;
  const int total = NL*per;
  for (int i = blockIdx.x*blockDim.x + threadIdx.x; i < total; i += gridDim.x*blockDim.x){
    int l = i / per, r = i % per;
    int k = r % KPH, n = r / KPH;
    int g = n / HP, j = n % HP;
    float v = 0.f;
    if (j < NH && k < NH) v = W[(size_t)l*NH*(5*NH) + (size_t)k*(5*NH) + g*NH + j]; // [8,300,1500]
    dst[i] = f2bf(v);
  }
}

__global__ void zero_f32_k(float* p, size_t n){
  for (size_t i = blockIdx.x*(size_t)blockDim.x + threadIdx.x; i < n;
       i += (size_t)gridDim.x*blockDim.x) p[i] = 0.f;
}
__global__ void init_recur_k(unsigned short* hbuf, int* bar){
  const int n = 2*NB*KPH;
  for (int i = blockIdx.x*blockDim.x + threadIdx.x; i < n; i += gridDim.x*blockDim.x)
    hbuf[i] = 0;
  if (blockIdx.x == 0 && threadIdx.x < 8) bar[threadIdx.x] = 0;
}

// ---------- stage layer input as bf16 [MR][kp], time-major rows (t*NB+b), optional reversal ----------
__global__ void stage_x_k(const float* __restrict__ src, const int* __restrict__ len,
                          unsigned short* __restrict__ xb,
                          int kp, int kvalid, int srcIsBTD, int rev){
  const size_t total = (size_t)MR*kp;
  for (size_t i = blockIdx.x*(size_t)blockDim.x + threadIdx.x; i < total;
       i += (size_t)gridDim.x*blockDim.x){
    int k = (int)(i % kp); int row = (int)(i / kp);
    int b = row % NB, t = row / NB;
    float v = 0.f;
    if (k < kvalid){
      int ts = t; bool ok = true;
      if (rev){ int L = len[b]; ts = L - 1 - t; ok = (ts >= 0); }
      if (ok){
        if (srcIsBTD) v = src[(size_t)b*NT*kvalid + (size_t)ts*kvalid + k]; // [B,T,D]
        else          v = src[((size_t)ts*NB + b)*(size_t)kvalid + k];      // [T,B,H]
      }
    }
    xb[i] = f2bf(v);
  }
}

// ---------- input projection: PI[MR][NG6] = XB[MR][KP] * WT[NG6][KP]^T ----------
// grid = (15, MR/16). Block stages its 16xKP A tile in LDS; each wave streams one
// 16-column B strip from L2. Fragment layouts per ISA 7.12.2:
//  A: m=lane%16, k in {h*8..+7, 16+h*8..+7}; B: n=lane%16, k in {h*8..+7, 16+h*8..+7}; C: m=r+8*h.
template<int KP>
__global__ void __launch_bounds__(256) proj_gemm_k(const __bf16* __restrict__ XB,
                                                   const __bf16* __restrict__ WT,
                                                   float* __restrict__ PI){
  __shared__ __align__(16) unsigned short sA[16][KP + LPAD];
  const int wave = threadIdx.x >> 5, lane = threadIdx.x & 31;
  const int ln16 = lane & 15, half = lane >> 4;
  const int NTI = NG6/16;                 // 114
  const int mt = blockIdx.y;
  // stage A tile (16 rows x KP) into LDS
  for (int idx = threadIdx.x; idx < 16*(KP/8); idx += 256){
    int row = idx / (KP/8), c = idx % (KP/8);
    *(v8bf*)&sA[row][c*8] = *(const v8bf*)(XB + (size_t)(mt*16 + row)*KP + c*8);
  }
  __syncthreads();
  int nt = blockIdx.x*8 + wave;
  if (nt < NTI){
    const unsigned short* Als = &sA[ln16][0];
    const __bf16* Brow = WT + (size_t)(nt*16 + ln16)*KP;
    v8f acc = {};
#pragma unroll
    for (int ks = 0; ks < KP/32; ++ks){
      v8bf a0 = *(const v8bf*)(Als + ks*32 + half*8);
      v8bf a1 = *(const v8bf*)(Als + ks*32 + 16 + half*8);
      v8bf b0 = *(const v8bf*)(Brow + ks*32 + half*8);
      v8bf b1 = *(const v8bf*)(Brow + ks*32 + 16 + half*8);
      acc = __builtin_amdgcn_wmma_f32_16x16x32_bf16(false, cat8(a0,a1), false, cat8(b0,b1),
                                                    (short)0, acc, false, false);
    }
#pragma unroll
    for (int r = 0; r < 8; ++r){
      int mg = mt*16 + r + 8*half;
      PI[(size_t)mg*NG6 + nt*16 + ln16] = acc[r];
    }
  }
}

// ---------- device-wide split barrier (persistent blocks) ----------
__device__ __forceinline__ void grid_barrier(int* bar, int nblk){
  __syncthreads();
  if (threadIdx.x == 0){
    __threadfence();
    int ph = __hip_atomic_load(&bar[1], __ATOMIC_RELAXED, __HIP_MEMORY_SCOPE_AGENT);
    int prev = __hip_atomic_fetch_add(&bar[0], 1, __ATOMIC_ACQ_REL, __HIP_MEMORY_SCOPE_AGENT);
    if (prev == nblk - 1){
      __hip_atomic_store(&bar[0], 0, __ATOMIC_RELAXED, __HIP_MEMORY_SCOPE_AGENT);
      __hip_atomic_store(&bar[1], ph + 1, __ATOMIC_RELEASE, __HIP_MEMORY_SCOPE_AGENT);
    } else {
      while (__hip_atomic_load(&bar[1], __ATOMIC_ACQUIRE, __HIP_MEMORY_SCOPE_AGENT) == ph)
        __builtin_amdgcn_s_sleep(1);
    }
  }
  __syncthreads();
}

// ---------- persistent recurrence: 19 blocks, one 16-wide hidden slice each ----------
// W_hh^T slice (80 rows x 320 K, bf16) lives in LDS for the whole layer; per step only
// h (40KB bf16) is exchanged through L2 + pi columns are read.
__global__ void __launch_bounds__(256) recurrence_k(const __bf16* __restrict__ WHH,
                                                    const float* __restrict__ bias,
                                                    const float* __restrict__ PI,
                                                    const int* __restrict__ len,
                                                    unsigned short* __restrict__ hbuf,
                                                    float* __restrict__ hid,   // [T,B,H] this layer
                                                    int rev, int* bar){
  __shared__ __align__(16) unsigned short sWB[80][KPH + LPAD]; // B slice, 5 gates x 16 cols
  __shared__ float sZ[5][NB][16];   // pre-activations, 5 gates
  __shared__ float sC[NB][16];      // cell state (f32, persistent)
  __shared__ float sH[NB][16];      // carried h  (f32, persistent)
  __shared__ int   sLen[NB];
  const int jblk = blockIdx.x;
  const int tid  = threadIdx.x;
  const int wave = tid >> 5, lane = tid & 31, ln16 = lane & 15, half = lane >> 4;

  // stage W_hh^T rows n = g*HP + jblk*16 + j  ->  sWB[g*16 + j][*]
  for (int idx = tid; idx < 80*(KPH/8); idx += 256){
    int row = idx / (KPH/8), c = idx % (KPH/8);
    int n = (row >> 4)*HP + jblk*16 + (row & 15);
    *(v8bf*)&sWB[row][c*8] = *(const v8bf*)(WHH + (size_t)n*KPH + c*8);
  }
  for (int e = tid; e < NB*16; e += 256){ sC[e>>4][e&15] = 0.f; sH[e>>4][e&15] = 0.f; }
  if (tid < NB) sLen[tid] = len[tid];
  __syncthreads();

  // wave w owns m-tile (w&3); gates {w>>2, w>>2 + 2, w>>2 + 4 (<5)}
  const int mt = wave & 3;
  const int g0 = wave >> 2;
  const int jcol = jblk*16 + ln16;

  for (int t = 0; t < NT; ++t){
    const __bf16* hcur = (const __bf16*)(hbuf + (t & 1)*(NB*KPH));
    unsigned short* hnxt = hbuf + ((t & 1) ^ 1)*(NB*KPH);

    // ---- preload A fragments (this wave's m-tile of h) once per step ----
    const __bf16* Arow = hcur + (size_t)(mt*16 + ln16)*KPH;
    v16bf Af[KPH/32];
#pragma unroll
    for (int ks = 0; ks < KPH/32; ++ks){
      v8bf a0 = *(const v8bf*)(Arow + ks*32 + half*8);
      v8bf a1 = *(const v8bf*)(Arow + ks*32 + 16 + half*8);
      Af[ks] = cat8(a0, a1);
    }

    // ---- GEMM phase: ps = h @ W_hh for this hidden slice (B from LDS) ----
    for (int g = g0; g < 5; g += 2){
      const unsigned short* Bls = &sWB[g*16 + ln16][0];
      v8f acc = {};
#pragma unroll
      for (int ks = 0; ks < KPH/32; ++ks){
        v8bf b0 = *(const v8bf*)(Bls + ks*32 + half*8);
        v8bf b1 = *(const v8bf*)(Bls + ks*32 + 16 + half*8);
        acc = __builtin_amdgcn_wmma_f32_16x16x32_bf16(false, Af[ks], false, cat8(b0,b1),
                                                      (short)0, acc, false, false);
      }
      float bb = (jcol < NH) ? bias[g*NH + jcol] : 0.f;
      const float* picol = PI + (size_t)(t*NB)*NG6 + g*HP + jblk*16 + ln16;
#pragma unroll
      for (int r = 0; r < 8; ++r){
        int ml = mt*16 + r + 8*half;                 // batch row within tile
        sZ[g][ml][ln16] = acc[r] + bb + picol[(size_t)ml*NG6];
      }
    }
    __syncthreads();

    // ---- elementwise gate phase ----
    for (int e = tid; e < NB*16; e += 256){
      int b = e >> 4, jl = e & 15;
      int jc = jblk*16 + jl;
      bool mk = (t < sLen[b]);
      float ig = sigm(sZ[0][b][jl]);
      float fg = sigm(sZ[1][b][jl]);
      float gg = tanhf(sZ[2][b][jl]);
      float og = sigm(sZ[3][b][jl]);
      float rg = sigm(sZ[4][b][jl]);
      float cold = sC[b][jl], hold = sH[b][jl];
      float cnew = ig*gg + fg*cold;
      float lstm = og * tanhf(cnew);
      float hw   = PI[(size_t)(t*NB + b)*NG6 + 5*HP + jc];
      float hnew = rg*lstm + (1.f - rg)*hw;
      float ccar = mk ? cnew : cold;
      float hcar = mk ? hnew : hold;
      sC[b][jl] = ccar; sH[b][jl] = hcar;
      hnxt[b*KPH + jc] = (jc < NH) ? f2bf(hcar) : (unsigned short)0;
      if (mk && jc < NH){
        int tout = rev ? (sLen[b] - 1 - t) : t;
        hid[((size_t)tout*NB + b)*NH + jc] = hnew;   // ys = h_new * mask (rest pre-zeroed)
      }
    }
    __threadfence();
    grid_barrier(bar, RBLK);
  }
}

// ---------- final [T,B,H] -> [B,T,H] ----------
__global__ void final_out_k(const float* __restrict__ hid7, float* __restrict__ out){
  const size_t n = (size_t)NB*NT*NH;
  for (size_t i = blockIdx.x*(size_t)blockDim.x + threadIdx.x; i < n;
       i += (size_t)gridDim.x*blockDim.x){
    int k = (int)(i % NH); int t = (int)((i / NH) % NT); int b = (int)(i / ((size_t)NH*NT));
    out[i] = hid7[((size_t)t*NB + b)*NH + k];
  }
}

static inline unsigned gsz(size_t n){
  size_t g = (n + 255) / 256; if (g > 4096) g = 4096; return (unsigned)g;
}

extern "C" void kernel_launch(void* const* d_in, const int* in_sizes, int n_in,
                              void* d_out, int out_size, void* d_ws, size_t ws_size,
                              hipStream_t stream){
  const float* inputs  = (const float*)d_in[0];   // [64,256,200]
  const int*   lengths = (const int*)  d_in[1];   // [64]
  const float* W_ih0   = (const float*)d_in[2];   // [200,1800]
  const float* W_ihr   = (const float*)d_in[3];   // [7,300,1800]
  const float* W_hh    = (const float*)d_in[4];   // [8,300,1500]
  const float* b_hh    = (const float*)d_in[5];   // [8,1500]
  float* out = (float*)d_out;                     // [B,T,H] ++ [L,T,B,H]

  char* ws = (char*)d_ws;
  size_t o = 0;
  unsigned short* WT0 = (unsigned short*)(ws + o); o += (size_t)NG6*KPD*2;
  unsigned short* WTR = (unsigned short*)(ws + o); o += (size_t)7*NG6*KPH*2;
  unsigned short* WTH = (unsigned short*)(ws + o); o += (size_t)NL*NG5*KPH*2;
  unsigned short* XB  = (unsigned short*)(ws + o); o += (size_t)MR*KPH*2;
  float*          PI  = (float*)(ws + o);         o += (size_t)MR*NG6*4;
  unsigned short* HB  = (unsigned short*)(ws + o);o += (size_t)2*NB*KPH*2;
  int*            BAR = (int*)(ws + o);           o += 256;

  float* hidden = out + (size_t)NB*NT*NH;         // [L,T,B,H]

  pack_wih0_k<<<gsz((size_t)NG6*KPD), 256, 0, stream>>>(W_ih0, WT0);
  pack_wihr_k<<<gsz((size_t)7*NG6*KPH), 256, 0, stream>>>(W_ihr, WTR);
  pack_whh_k <<<gsz((size_t)NL*NG5*KPH), 256, 0, stream>>>(W_hh, WTH);
  zero_f32_k <<<gsz((size_t)out_size), 256, 0, stream>>>(out, (size_t)out_size);

  const dim3 projGrid(15, MR/16, 1);              // 15 n-groups x 1024 m-tiles

  for (int l = 0; l < NL; ++l){
    int rev = l & 1;
    init_recur_k<<<gsz(2*NB*KPH), 256, 0, stream>>>(HB, BAR);
    if (l == 0){
      stage_x_k<<<gsz((size_t)MR*KPD), 256, 0, stream>>>(inputs, lengths, XB,
                                                         KPD, ND, 1, 0);
      proj_gemm_k<KPD><<<projGrid, 256, 0, stream>>>((const __bf16*)XB,
                                                     (const __bf16*)WT0, PI);
    } else {
      stage_x_k<<<gsz((size_t)MR*KPH), 256, 0, stream>>>(hidden + (size_t)(l-1)*NT*NB*NH,
                                                         lengths, XB, KPH, NH, 0, rev);
      proj_gemm_k<KPH><<<projGrid, 256, 0, stream>>>((const __bf16*)XB,
          (const __bf16*)(WTR + (size_t)(l-1)*NG6*KPH), PI);
    }
    recurrence_k<<<RBLK, 256, 0, stream>>>((const __bf16*)(WTH + (size_t)l*NG5*KPH),
                                           b_hh + (size_t)l*5*NH, PI, lengths, HB,
                                           hidden + (size_t)l*NT*NB*NH, rev, BAR);
  }
  final_out_k<<<gsz((size_t)NB*NT*NH), 256, 0, stream>>>(hidden + (size_t)7*NT*NB*NH, out);
}
